// PGAS_4690104287503
// MI455X (gfx1250) — compile-verified, matrix operand order checked by reading, not run on hardware
//
#include <hip/hip_runtime.h>
#include <hip/hip_bf16.h>
#include <math.h>

#define T_STEPS 400
#define NP      4096
#define DXD     32
#define DUD     16
#define DBD     1024
#define DZ      64          // K of GEMM1, padded 48 -> 64
#define NCHUNK  (DBD / 32)  // 32 K-chunks over the basis dimension
#define SIGMA_Y_F 0.1f
#define PI_F 3.14159265358979323846f

typedef __attribute__((ext_vector_type(16))) _Float16 v16h;
typedef __attribute__((ext_vector_type(8)))  float    v8f;

// ---------------------------------------------------------------------------
// WMMA helpers (CDNA5 wave32: 16x16x32 f16 -> f32)
// 16-bit A/B fragment layout (ISA 7.12.2): lane l covers M (or N) = l&15;
// h = l>>4 selects K-subblock {0..7,16..23} (h=0) vs {8..15,24..31} (h=1).
// ---------------------------------------------------------------------------
__device__ __forceinline__ int frag_k(int h, int i) {
  return (i < 8) ? (h * 8 + i) : (16 + h * 8 + (i - 8));
}

__device__ __forceinline__ v16h ldfrag_lds(const _Float16* p, int h) {
  v16h r;
#pragma unroll
  for (int i = 0; i < 8; ++i)  r[i]     = p[h * 8 + i];
#pragma unroll
  for (int i = 0; i < 8; ++i)  r[8 + i] = p[16 + h * 8 + i];
  return r;
}

__device__ __forceinline__ v8f wmma16(v16h a, v16h b, v8f c) {
  return __builtin_amdgcn_wmma_f32_16x16x32_f16(
      /*neg_a=*/false, a, /*neg_b=*/false, b,
      /*c_mod=*/(short)0, c, /*reuse_a=*/false, /*reuse_b=*/false);
}

// Branch-free tanh: hardware v_tanh_f32 when available, else v_exp_f32 based.
__device__ __forceinline__ float fast_tanh(float x) {
#if __has_builtin(__builtin_amdgcn_tanhf)
  return __builtin_amdgcn_tanhf(x);
#else
  const float xc = fminf(fmaxf(x, -10.f), 10.f);
  const float t  = __builtin_amdgcn_exp2f(2.8853900817779268f * xc); // 2*log2(e)*x
  return (t - 1.f) * __builtin_amdgcn_rcpf(t + 1.f);
#endif
}

__device__ __forceinline__ int lower_bound_dev(const float* a, int n, float v) {
  int lo = 0, hi = n;
  while (lo < hi) {
    int mid = (lo + hi) >> 1;
    if (a[mid] < v) lo = mid + 1; else hi = mid;
  }
  return lo;
}

// ---------------------------------------------------------------------------
// Fused per-step kernel:  basis = tanh(z @ W^T) ; aux = basis @ C^T
// One wave owns a 16-row particle tile. GEMM1 produces basis in 16x32 chunks
// (two 16x16 WMMA tiles, K=64 padded), tanh'd to f16 via LDS relayout, then
// immediately consumed as the A-fragment of GEMM2 (K accumulation over DB).
// Weights arrive pre-swizzled in fragment-native order -> pure b128 traffic.
// ---------------------------------------------------------------------------
__global__ __launch_bounds__(256)
void k_fused_basis_aux(const float* __restrict__ state,   // NP x DXD
                       const float* __restrict__ u_t,     // DUD
                       const _Float16* __restrict__ Whf,  // NCHUNK x 4 x 32 x 16
                       const _Float16* __restrict__ Chf,  // NCHUNK x 2 x 32 x 16
                       float* __restrict__ aux)           // NP x DXD
{
  __shared__ _Float16 zh[8][16][DZ];    // z tile per wave (f16, K padded)
  __shared__ _Float16 bch[8][16][32];   // basis chunk per wave

  const int wave = threadIdx.x >> 5;
  const int lane = threadIdx.x & 31;
  const int rowbase = (blockIdx.x * 8 + wave) * 16;

  // Stage z = [state | u_t | 0-pad] for this wave's 16 rows.
  if (lane < 16) {
    const float* srow = state + (size_t)(rowbase + lane) * DXD;
#pragma unroll 4
    for (int k = 0; k < DXD; ++k) zh[wave][lane][k] = (_Float16)srow[k];
#pragma unroll
    for (int k = 0; k < DUD; ++k) zh[wave][lane][DXD + k] = (_Float16)u_t[k];
#pragma unroll
    for (int k = DXD + DUD; k < DZ; ++k) zh[wave][lane][k] = (_Float16)0.f;
  }
  asm volatile("s_wait_dscnt 0" ::: "memory");   // per-wave LDS store->load order

  const int m = lane & 15;
  const int h = lane >> 4;
  const v16h a0 = ldfrag_lds(&zh[wave][m][0],  h);   // z K = 0..31
  const v16h a1 = ldfrag_lds(&zh[wave][m][32], h);   // z K = 32..63

  const v8f zero = {0.f, 0.f, 0.f, 0.f, 0.f, 0.f, 0.f, 0.f};
  v8f acc0 = zero;   // aux cols 0..15
  v8f acc1 = zero;   // aux cols 16..31

#pragma unroll 2
  for (int c = 0; c < NCHUNK; ++c) {
    const _Float16* wf = Whf + ((size_t)c * 4 * 32 + lane) * 16;
    const _Float16* cf = Chf + ((size_t)c * 2 * 32 + lane) * 16;
    if (c + 1 < NCHUNK)
      __builtin_prefetch(wf + (size_t)4 * 32 * 16, 0, 1);  // global_prefetch_b8

    // GEMM1: basis chunk cols 32c..32c+31 (fragment-native b128 loads).
    const v16h bw00 = *(const v16h*)(wf);                // tile0, K 0..31
    const v16h bw01 = *(const v16h*)(wf + 32 * 16);      // tile0, K 32..63
    const v16h bw10 = *(const v16h*)(wf + 2 * 32 * 16);  // tile1, K 0..31
    const v16h bw11 = *(const v16h*)(wf + 3 * 32 * 16);  // tile1, K 32..63
    v8f c0 = zero, c1 = zero;
    c0 = wmma16(a0, bw00, c0);
    c0 = wmma16(a1, bw01, c0);
    c1 = wmma16(a0, bw10, c1);
    c1 = wmma16(a1, bw11, c1);

    // Hardware tanh, downconvert, park 16x32 chunk in LDS
    // (C layout: row = v + 8h, col = m / 16+m).
#pragma unroll
    for (int v = 0; v < 8; ++v) {
      bch[wave][v + 8 * h][m]      = (_Float16)fast_tanh(c0[v]);
      bch[wave][v + 8 * h][16 + m] = (_Float16)fast_tanh(c1[v]);
    }
    asm volatile("s_wait_dscnt 0" ::: "memory");

    // GEMM2: aux += basis_chunk(16x32) @ C_chunk(32x32).
    const v16h a2  = ldfrag_lds(&bch[wave][m][0], h);
    const v16h cb0 = *(const v16h*)(cf);
    const v16h cb1 = *(const v16h*)(cf + 32 * 16);
    acc0 = wmma16(a2, cb0, acc0);
    acc1 = wmma16(a2, cb1, acc1);
  }

#pragma unroll
  for (int v = 0; v < 8; ++v) {
    const int r = rowbase + v + 8 * h;
    aux[(size_t)r * DXD + m]      = acc0[v];
    aux[(size_t)r * DXD + 16 + m] = acc1[v];
  }
}

// ---------------------------------------------------------------------------
// Per-particle stats: ll_aux, lw_aux = ll_aux + lw, lw_anc = lw_aux + h_x
// h_x uses precomputed L^{-1} (lower-triangular) and sum(log diag L).
// ---------------------------------------------------------------------------
__global__ __launch_bounds__(256)
void k_particle_stats(const float* __restrict__ aux,
                      const float* __restrict__ y_t,
                      const float* __restrict__ ref_t,
                      const float* __restrict__ Linv,    // DXD x DXD
                      const float* __restrict__ consts,  // [0] = sum log diag L
                      const float* __restrict__ lw,
                      float* __restrict__ ll_aux,
                      float* __restrict__ lw_aux,
                      float* __restrict__ lw_anc)
{
  const int i = blockIdx.x * blockDim.x + threadIdx.x;
  if (i >= NP) return;
  const float* a = aux + (size_t)i * DXD;

  float d2 = 0.f;
  float dif[DXD];
#pragma unroll
  for (int k = 0; k < DXD; ++k) {
    const float ak = a[k];
    const float dd = y_t[k] - ak;
    d2 += dd * dd;
    dif[k] = ref_t[k] - ak;
  }
  const float inv2s2 = 0.5f / (SIGMA_Y_F * SIGMA_Y_F);
  const float cl = 0.5f * (float)DXD * logf(2.f * PI_F * SIGMA_Y_F * SIGMA_Y_F);
  const float ll = -d2 * inv2s2 - cl;

  float z2 = 0.f;
  for (int r = 0; r < DXD; ++r) {
    float s = 0.f;
    for (int c = 0; c <= r; ++c) s += Linv[r * DXD + c] * dif[c];
    z2 += s * s;
  }
  const float hx = -0.5f * z2 - consts[0] - 0.5f * (float)DXD * logf(2.f * PI_F);

  ll_aux[i] = ll;
  const float la = ll + lw[i];
  lw_aux[i] = la;
  lw_anc[i] = la + hx;
}

// ---------------------------------------------------------------------------
// Single-workgroup softmax + inclusive scan (4096) -> cdf in shared memory.
// Returns total mass; comparisons use v = pos * total (avoids the divide).
// ---------------------------------------------------------------------------
__device__ float build_cdf(const float* __restrict__ lwv, float* sh, float* red) {
  const int tid = threadIdx.x;
  float loc[4];
  float m = -INFINITY;
#pragma unroll
  for (int j = 0; j < 4; ++j) { loc[j] = lwv[4 * tid + j]; m = fmaxf(m, loc[j]); }
  red[tid] = m;
  __syncthreads();
  for (int off = 512; off > 0; off >>= 1) {
    if (tid < off) red[tid] = fmaxf(red[tid], red[tid + off]);
    __syncthreads();
  }
  m = red[0];
  __syncthreads();

  float s = 0.f, pre[4];
#pragma unroll
  for (int j = 0; j < 4; ++j) { s += expf(loc[j] - m); pre[j] = s; }
  red[tid] = s;
  __syncthreads();
  for (int off = 1; off < 1024; off <<= 1) {        // Hillis-Steele inclusive
    float v = (tid >= off) ? red[tid - off] : 0.f;
    __syncthreads();
    red[tid] += v;
    __syncthreads();
  }
  const float base = red[tid] - s;
#pragma unroll
  for (int j = 0; j < 4; ++j) sh[4 * tid + j] = base + pre[j];
  const float total = red[1023];
  __syncthreads();
  return total;
}

// Systematic resampling + ancestor index for the reference trajectory.
__global__ __launch_bounds__(1024)
void k_resample(const float* __restrict__ lw_aux,
                const float* __restrict__ lw_anc,
                const float* __restrict__ u_sys_t,
                const float* __restrict__ u_anc_t,
                int* __restrict__ a_idx)
{
  __shared__ float sh[NP];
  __shared__ float red[1024];
  const int tid = threadIdx.x;

  float total = build_cdf(lw_aux, sh, red);
  const float us = u_sys_t[0];
#pragma unroll
  for (int j = 0; j < 4; ++j) {
    const int i = 4 * tid + j;
    const float v = ((float)i + us) * (1.0f / (float)NP) * total;
    int a = lower_bound_dev(sh, NP, v);
    a_idx[i] = (a < NP - 1) ? a : (NP - 1);
  }
  __syncthreads();

  total = build_cdf(lw_anc, sh, red);
  if (tid == 0) {
    int r = lower_bound_dev(sh, NP, u_anc_t[0] * total);
    a_idx[NP - 1] = (r < NP - 1) ? r : (NP - 1);
  }
}

// ---------------------------------------------------------------------------
// Propagate: new_state = aux + eps @ L^T (row NP-1 := ref_t),
// lw = loglik(y, new_state) - ll_aux[a_idx]; record state + trace.
// ---------------------------------------------------------------------------
__global__ __launch_bounds__(256)
void k_propagate(const float* __restrict__ aux,
                 const float* __restrict__ noise_t,  // NP x DXD
                 const float* __restrict__ ref_t,
                 const float* __restrict__ y_t,
                 const float* __restrict__ Lmat,     // DXD x DXD lower
                 const float* __restrict__ ll_aux,
                 const int* __restrict__ a_idx,
                 float* __restrict__ state,
                 float* __restrict__ lw,
                 float* __restrict__ strace_t)
{
  const int i = blockIdx.x * blockDim.x + threadIdx.x;
  if (i >= NP) return;
  const float* e = noise_t + (size_t)i * DXD;
  const float* a = aux + (size_t)i * DXD;

  float ns[DXD];
  if (i == NP - 1) {
#pragma unroll
    for (int d = 0; d < DXD; ++d) ns[d] = ref_t[d];
  } else {
    for (int d = 0; d < DXD; ++d) {
      float s = a[d];
      for (int k = 0; k <= d; ++k) s += e[k] * Lmat[d * DXD + k];  // L lower-tri
      ns[d] = s;
    }
  }

  float d2 = 0.f;
#pragma unroll
  for (int d = 0; d < DXD; ++d) { const float dd = y_t[d] - ns[d]; d2 += dd * dd; }
  const float inv2s2 = 0.5f / (SIGMA_Y_F * SIGMA_Y_F);
  const float cl = 0.5f * (float)DXD * logf(2.f * PI_F * SIGMA_Y_F * SIGMA_Y_F);
  lw[i] = (-d2 * inv2s2 - cl) - ll_aux[a_idx[i]];

#pragma unroll
  for (int d = 0; d < DXD; ++d) {
    state[(size_t)i * DXD + d]    = ns[d];
    strace_t[(size_t)i * DXD + d] = ns[d];
  }
}

// ---------------------------------------------------------------------------
// One-time setup kernels.
// ---------------------------------------------------------------------------
__global__ void k_init(const float* __restrict__ init_states,
                       const float* __restrict__ ref0,
                       float* __restrict__ state,
                       float* __restrict__ strace0,
                       float* __restrict__ lw)
{
  const int i = blockIdx.x * blockDim.x + threadIdx.x;
  if (i >= NP) return;
  lw[i] = 0.f;
  for (int d = 0; d < DXD; ++d) {
    const float v = (i == NP - 1) ? ref0[d] : init_states[(size_t)i * DXD + d];
    state[(size_t)i * DXD + d]   = v;
    strace0[(size_t)i * DXD + d] = v;
  }
}

// Pre-swizzle both weight matrices into WMMA B-fragment-native order.
// Whf[(c*4 + f)*32 + lane][i]: f = tile*2 + khalf; value = W[32c+16*tile+(l&15)][32*khalf + K(h,i)]
// Chf[(c*2 + t)*32 + lane][i]: value = coeff[(16t+(l&15)) * DBD + 32c + K(h,i)]
__global__ void k_swizzle(const float* __restrict__ W,     // DBD x 48
                          const float* __restrict__ Cf,    // DXD x DBD
                          _Float16* __restrict__ Whf,      // NCHUNK*4*32*16
                          _Float16* __restrict__ Chf)      // NCHUNK*2*32*16
{
  const int idx = blockIdx.x * blockDim.x + threadIdx.x;
  if (idx < NCHUNK * 4 * 32 * 16) {
    const int i    = idx & 15;
    const int l    = (idx >> 4) & 31;
    const int f    = (idx >> 9) & 3;
    const int c    = idx >> 11;
    const int tile = f >> 1, khalf = f & 1;
    const int h    = l >> 4;
    const int K    = 32 * khalf + frag_k(h, i);
    const int row  = c * 32 + tile * 16 + (l & 15);
    Whf[idx] = (K < DXD + DUD) ? (_Float16)W[(size_t)row * (DXD + DUD) + K]
                               : (_Float16)0.f;
  }
  if (idx < NCHUNK * 2 * 32 * 16) {
    const int i    = idx & 15;
    const int l    = (idx >> 4) & 31;
    const int tile = (idx >> 9) & 1;
    const int c    = idx >> 10;
    const int h    = l >> 4;
    const int K    = frag_k(h, i);
    const int n    = tile * 16 + (l & 15);
    Chf[idx] = (_Float16)Cf[(size_t)n * DBD + c * 32 + K];
  }
}

// Cholesky of 32x32 error_cov + forward-substitution inverse, single thread.
__global__ void k_precompute(const float* __restrict__ ecov,
                             float* __restrict__ Lmat,
                             float* __restrict__ Linv,
                             float* __restrict__ consts)
{
  if (threadIdx.x != 0 || blockIdx.x != 0) return;
  float L[DXD][DXD];
  for (int r = 0; r < DXD; ++r)
    for (int c = 0; c < DXD; ++c) L[r][c] = 0.f;
  for (int j = 0; j < DXD; ++j) {
    float s = ecov[j * DXD + j];
    for (int k = 0; k < j; ++k) s -= L[j][k] * L[j][k];
    L[j][j] = sqrtf(s);
    for (int i = j + 1; i < DXD; ++i) {
      float t = ecov[i * DXD + j];
      for (int k = 0; k < j; ++k) t -= L[i][k] * L[j][k];
      L[i][j] = t / L[j][j];
    }
  }
  float sld = 0.f;
  for (int j = 0; j < DXD; ++j) sld += logf(L[j][j]);
  consts[0] = sld;

  float Li[DXD][DXD];
  for (int r = 0; r < DXD; ++r)
    for (int c = 0; c < DXD; ++c) Li[r][c] = 0.f;
  for (int j = 0; j < DXD; ++j) {
    Li[j][j] = 1.f / L[j][j];
    for (int i = j + 1; i < DXD; ++i) {
      float s = 0.f;
      for (int k = j; k < i; ++k) s += L[i][k] * Li[k][j];
      Li[i][j] = -s / L[i][i];
    }
  }
  for (int r = 0; r < DXD; ++r)
    for (int c = 0; c < DXD; ++c) {
      Lmat[r * DXD + c] = L[r][c];
      Linv[r * DXD + c] = Li[r][c];
    }
}

// ---------------------------------------------------------------------------
// Finalize: softmax over final lw -> draw idx -> backward ancestor trace.
// ---------------------------------------------------------------------------
__global__ __launch_bounds__(1024)
void k_finalize(const float* __restrict__ lw,
                const float* __restrict__ u_final,
                const int* __restrict__ anc,       // (T-1) x NP
                const float* __restrict__ strace,  // T x NP x DXD
                float* __restrict__ out)           // T x DXD
{
  __shared__ float sh[NP];
  __shared__ float red[1024];
  __shared__ int bseq[T_STEPS];
  const int tid = threadIdx.x;

  const float total = build_cdf(lw, sh, red);
  if (tid == 0) {
    int idx = lower_bound_dev(sh, NP, u_final[0] * total);
    idx = (idx < NP - 1) ? idx : (NP - 1);
    bseq[T_STEPS - 1] = idx;
    int b = idx;
    for (int t = T_STEPS - 2; t >= 0; --t) {
      b = anc[(size_t)t * NP + b];
      bseq[t] = b;
    }
  }
  __syncthreads();

  for (int e = tid; e < T_STEPS * DXD; e += 1024) {
    const int t = e / DXD, d = e % DXD;
    out[e] = strace[(size_t)t * NP * DXD + (size_t)bseq[t] * DXD + d];
  }
}

// ---------------------------------------------------------------------------
// Host driver (graph-capture safe: only kernel launches on `stream`).
// ---------------------------------------------------------------------------
extern "C" void kernel_launch(void* const* d_in, const int* in_sizes, int n_in,
                              void* d_out, int out_size, void* d_ws, size_t ws_size,
                              hipStream_t stream) {
  (void)in_sizes; (void)n_in; (void)out_size; (void)ws_size;
  const float* obs   = (const float*)d_in[0];   // T x DXD
  const float* inp   = (const float*)d_in[1];   // T x DUD
  const float* ist   = (const float*)d_in[2];   // NP x DXD
  const float* ref   = (const float*)d_in[3];   // T x DXD
  const float* coeff = (const float*)d_in[4];   // DXD x DBD
  const float* ecov  = (const float*)d_in[5];   // DXD x DXD
  const float* Wb    = (const float*)d_in[6];   // DBD x 48
  const float* noise = (const float*)d_in[7];   // T x NP x DXD
  const float* usys  = (const float*)d_in[8];   // T
  const float* uanc  = (const float*)d_in[9];   // T
  const float* ufin  = (const float*)d_in[10];  // 1

  // Carve workspace (256B aligned slices).
  char* p = (char*)d_ws;
  auto carve = [&](size_t bytes) {
    char* r = p;
    p += (bytes + 255) & ~(size_t)255;
    return r;
  };
  float*     Lmat   = (float*)carve(DXD * DXD * sizeof(float));
  float*     Linv   = (float*)carve(DXD * DXD * sizeof(float));
  float*     consts = (float*)carve(16 * sizeof(float));
  _Float16*  Whf    = (_Float16*)carve((size_t)NCHUNK * 4 * 32 * 16 * sizeof(_Float16));
  _Float16*  Chf    = (_Float16*)carve((size_t)NCHUNK * 2 * 32 * 16 * sizeof(_Float16));
  float*     state  = (float*)carve((size_t)NP * DXD * sizeof(float));
  float*     aux    = (float*)carve((size_t)NP * DXD * sizeof(float));
  float*     lw     = (float*)carve(NP * sizeof(float));
  float*     ll_aux = (float*)carve(NP * sizeof(float));
  float*     lw_aux = (float*)carve(NP * sizeof(float));
  float*     lw_anc = (float*)carve(NP * sizeof(float));
  int*       anc    = (int*)carve((size_t)(T_STEPS - 1) * NP * sizeof(int));
  float*     strace = (float*)carve((size_t)T_STEPS * NP * DXD * sizeof(float));

  k_precompute<<<1, 1, 0, stream>>>(ecov, Lmat, Linv, consts);
  k_swizzle<<<(NCHUNK * 4 * 32 * 16 + 255) / 256, 256, 0, stream>>>(Wb, coeff, Whf, Chf);
  k_init<<<(NP + 255) / 256, 256, 0, stream>>>(ist, ref, state, strace, lw);

  for (int t = 1; t < T_STEPS; ++t) {
    k_fused_basis_aux<<<NP / 128, 256, 0, stream>>>(
        state, inp + (size_t)t * DUD, Whf, Chf, aux);
    k_particle_stats<<<NP / 256, 256, 0, stream>>>(
        aux, obs + (size_t)t * DXD, ref + (size_t)t * DXD,
        Linv, consts, lw, ll_aux, lw_aux, lw_anc);
    k_resample<<<1, 1024, 0, stream>>>(
        lw_aux, lw_anc, usys + t, uanc + t, anc + (size_t)(t - 1) * NP);
    k_propagate<<<NP / 256, 256, 0, stream>>>(
        aux, noise + (size_t)t * NP * DXD, ref + (size_t)t * DXD,
        obs + (size_t)t * DXD, Lmat, ll_aux, anc + (size_t)(t - 1) * NP,
        state, lw, strace + (size_t)t * NP * DXD);
  }

  k_finalize<<<1, 1024, 0, stream>>>(lw, ufin, anc, strace, (float*)d_out);
}
